// OSGDecoder_4844723110283
// MI455X (gfx1250) — matrix operand order repaired and checked by reading, not compiled, not run on hardware
//
#include <hip/hip_runtime.h>

// ---------------------------------------------------------------------------
// Fused OSGDecoder for MI455X (gfx1250, wave32, WMMA f32_16x16x32_f16).
// One pass: features -> fp16 mean -> [tex FC64 -> softplus -> FC32 -> sigmoid]
//                               and [5x (FC64 -> softplus -> dot64)] -> sdf act.
// Memory-bound (~0.5 GB moved, ~15 GFLOP): fully fused, f16 WMMA for all GEMMs,
// non-temporal hints on the single-use feature/output streams, prefetch for
// the next tile, weights converted once per block into LDS.
// ---------------------------------------------------------------------------

typedef __attribute__((ext_vector_type(16))) _Float16 v16h;
typedef __attribute__((ext_vector_type(8)))  float    v8f;
typedef __attribute__((ext_vector_type(4)))  float    fv4;   // native ext vector
                                                             // (HIP float4 is a
                                                             // class; NT builtin
                                                             // rejects it)

#define N_BATCH 4
#define M_DIM   131072
#define C_DIM   32
#define NM      (N_BATCH * M_DIM)         /* 524288 points */
#define NTILES  (NM / 16)                 /* 32768 16-point tiles */
#define BLOCKS  1024
#define WPB     8                         /* waves per block */
#define TPW     (NTILES / (BLOCKS * WPB)) /* 4 tiles per wave */

#define L1P 40   /* LDS pitch in halves for layer-1 weight rows (32 cols used) */
#define HP  72   /* LDS pitch in halves for 64-wide rows (H staging, W_tex2)   */

union U16 { uint4 q[2]; v16h v; _Float16 e[16]; };
union F4  { fv4 q; float e[4]; };

__device__ __forceinline__ float softplus_f(float x) {
  return fmaxf(x, 0.0f) + __logf(1.0f + __expf(-fabsf(x)));
}

__device__ __forceinline__ v8f wmma_f16(v16h a, v16h b, v8f c) {
  // (neg_a, A, neg_b, B, c_mod, C, reuse_a, reuse_b)
  return __builtin_amdgcn_wmma_f32_16x16x32_f16(false, a, false, b, (short)0, c,
                                                false, false);
}

// Load one plane's 16 channels for this lane (A-fragment order: halves 0..7 are
// K = 8*h2+i, halves 8..15 are K = 16+8*h2+i) and convert to packed f16.
// Non-temporal: features are single-use streaming data.
__device__ __forceinline__ v16h load_plane_h16(const float* __restrict__ base,
                                               int h2) {
  F4 q0, q1, q2, q3;
  q0.q = __builtin_nontemporal_load((const fv4*)(base + h2 * 8));
  q1.q = __builtin_nontemporal_load((const fv4*)(base + h2 * 8) + 1);
  q2.q = __builtin_nontemporal_load((const fv4*)(base + 16 + h2 * 8));
  q3.q = __builtin_nontemporal_load((const fv4*)(base + 16 + h2 * 8) + 1);
  U16 u;
#pragma unroll
  for (int i = 0; i < 4; ++i) {
    u.e[i]      = (_Float16)q0.e[i];
    u.e[4 + i]  = (_Float16)q1.e[i];
    u.e[8 + i]  = (_Float16)q2.e[i];
    u.e[12 + i] = (_Float16)q3.e[i];
  }
  return u.v;
}

// fp16 mean over the 3 planes, entirely in packed-f16 ALU (v_pk_add/mul_f16).
__device__ __forceinline__ v16h load_feat_A(const float* __restrict__ f,
                                            int n, int row, int h2) {
  const size_t ps = (size_t)M_DIM * C_DIM;
  const float* b0 = f + ((size_t)(n * 3 + 0) * M_DIM + row) * C_DIM;
  v16h a0 = load_plane_h16(b0, h2);
  v16h a1 = load_plane_h16(b0 + ps, h2);
  v16h a2 = load_plane_h16(b0 + 2 * ps, h2);
  const _Float16 third = (_Float16)(1.0f / 3.0f);
  return (a0 + a1 + a2) * third;
}

__global__ __launch_bounds__(256, 2)
void osg_decoder_kernel(const float* __restrict__ tex,
                        const float* __restrict__ shp,
                        const float* __restrict__ Wtex1, const float* __restrict__ btex1,
                        const float* __restrict__ Wtex2, const float* __restrict__ btex2,
                        const float* __restrict__ Wshp1, const float* __restrict__ bshp1,
                        const float* __restrict__ Wshp2, const float* __restrict__ bshp2,
                        const float* __restrict__ Waux1, const float* __restrict__ baux1,
                        const float* __restrict__ Waux2, const float* __restrict__ baux2,
                        float* __restrict__ out) {
  // Layer-1 weights (tex1, shp1, aux1[0..3]) as gain-scaled halves, padded pitch.
  __shared__ alignas(16) _Float16 sW1[6 * 64 * L1P];
  __shared__ alignas(16) _Float16 sW2[32 * HP];       // W_tex2 * g2, [o][h]
  __shared__ float sB1[6 * 64];                       // layer-1 biases
  __shared__ float sB2t[32];                          // b_tex2
  __shared__ float sW2s[5 * 64];                      // (W_shp2 | W_aux2[k]) * g2
  __shared__ float sB2s[8];                           // b_shp2, b_aux2[0..3]
  __shared__ alignas(16) _Float16 sH[WPB][16 * HP];   // per-wave H staging

  const int tid = threadIdx.x;
  const float g1 = 0.17677669529663687f;  // 1/sqrt(32)
  const float g2 = 0.125f;                // 1/sqrt(64)

  // ---- one-time per-block weight conversion into LDS ----
  for (int i = tid; i < 6 * 64 * 32; i += 256) {
    int mat = i >> 11, rem = i & 2047;
    float w = (mat == 0) ? Wtex1[rem]
            : (mat == 1) ? Wshp1[rem]
            :              Waux1[(mat - 2) * 2048 + rem];
    sW1[(mat * 64 + (rem >> 5)) * L1P + (rem & 31)] = (_Float16)(w * g1);
  }
  for (int i = tid; i < 32 * 64; i += 256)
    sW2[(i >> 6) * HP + (i & 63)] = (_Float16)(Wtex2[i] * g2);
  for (int i = tid; i < 6 * 64; i += 256) {
    int mat = i >> 6, h = i & 63;
    sB1[i] = (mat == 0) ? btex1[h] : (mat == 1) ? bshp1[h] : baux1[(mat - 2) * 64 + h];
  }
  if (tid < 32) sB2t[tid] = btex2[tid];
  for (int i = tid; i < 5 * 64; i += 256) {
    int g = i >> 6, h = i & 63;
    sW2s[i] = ((g == 0) ? Wshp2[h] : Waux2[(g - 1) * 64 + h]) * g2;
  }
  if (tid < 5) sB2s[tid] = (tid == 0) ? bshp2[0] : baux2[tid - 1];
  __syncthreads();

  const int wave = tid >> 5;
  const int lane = tid & 31;
  const int m16  = lane & 15;
  const int h2   = lane >> 4;
  _Float16* Hst = &sH[wave][0];

  float* o_rgb  = out;
  float* o_sig  = out + (size_t)NM * 32;
  float* o_sdf  = o_sig + NM;
  float* o_asig = o_sdf + NM;
  float* o_asdf = o_asig + (size_t)NM * 4;

  const v8f zero = {0.f, 0.f, 0.f, 0.f, 0.f, 0.f, 0.f, 0.f};
  const float inv_d = 1.0f / 0.1001f;  // 1/(|BETA|+1e-4)

  const int gwave = blockIdx.x * WPB + wave;
  for (int it = 0; it < TPW; ++it) {
    const int tile = gwave * TPW + it;
    const int p0 = tile * 16;
    const int n = p0 >> 17;                     // p0 / M_DIM
    const int mrow = (p0 & (M_DIM - 1)) + m16;  // row within this batch entry

    // Prefetch next tile's feature lines (global_prefetch_b8, speculative RT).
    if (it + 1 < TPW) {
      const int p0n = p0 + 16;
      const int nn = p0n >> 17;
      const int mrn = (p0n & (M_DIM - 1)) + m16;
#pragma unroll
      for (int pl = 0; pl < 3; ++pl) {
        const size_t off = ((size_t)(nn * 3 + pl) * M_DIM + mrn) * C_DIM + h2 * 8;
        __builtin_prefetch(tex + off, 0, 0);
        __builtin_prefetch(shp + off, 0, 0);
      }
    }

    v16h aT = load_feat_A(tex, n, mrow, h2);
    v16h aS = load_feat_A(shp, n, mrow, h2);

    // ================= texture branch =================
    v8f hT[4];
#pragma unroll
    for (int t = 0; t < 4; ++t) {
      U16 ub;
      const uint4* pw = (const uint4*)(sW1 + ((t * 16 + m16) * L1P + h2 * 8));
      ub.q[0] = pw[0]; ub.q[1] = pw[2];
      hT[t] = wmma_f16(aT, ub.v, zero);
    }
    // bias + softplus, stage as fp16 [m][h] for the layer-2 A transpose
#pragma unroll
    for (int t = 0; t < 4; ++t) {
      float bb = sB1[t * 16 + m16];
#pragma unroll
      for (int r = 0; r < 8; ++r)
        Hst[(r + 8 * h2) * HP + t * 16 + m16] = (_Float16)softplus_f(hT[t][r] + bb);
    }
    asm volatile("s_wait_dscnt 0" ::: "memory");  // wave's LDS stores visible

#pragma unroll
    for (int nt = 0; nt < 2; ++nt) {
      v8f acc = zero;
#pragma unroll
      for (int c2 = 0; c2 < 2; ++c2) {
        U16 ua, ub;
        const uint4* pa0 = (const uint4*)(Hst + m16 * HP + c2 * 32 + h2 * 8);
        const uint4* pa1 = (const uint4*)(Hst + m16 * HP + c2 * 32 + 16 + h2 * 8);
        ua.q[0] = pa0[0]; ua.q[1] = pa1[0];
        const uint4* pw = (const uint4*)(sW2 + (nt * 16 + m16) * HP + c2 * 32 + h2 * 8);
        ub.q[0] = pw[0]; ub.q[1] = pw[2];
        acc = wmma_f16(ua.v, ub.v, acc);
      }
      float bo = sB2t[nt * 16 + m16];
#pragma unroll
      for (int r = 0; r < 8; ++r) {
        float z = acc[r] + bo;
        float sg = 1.0f / (1.0f + __expf(-z));
        __builtin_nontemporal_store(
            sg * 1.002f - 0.001f,
            o_rgb + (size_t)(p0 + r + 8 * h2) * 32 + nt * 16 + m16);
      }
    }

    // ============ shape branch: g=0 main sdf, g=1..4 aux ============
    float keep[8];
#pragma unroll
    for (int r = 0; r < 8; ++r) keep[r] = 0.f;

#pragma unroll
    for (int g = 0; g < 5; ++g) {
      const int mat = g + 1;
      v8f hg[4];
#pragma unroll
      for (int t = 0; t < 4; ++t) {
        U16 ub;
        const uint4* pw = (const uint4*)(sW1 + ((mat * 64 + t * 16 + m16) * L1P + h2 * 8));
        ub.q[0] = pw[0]; ub.q[1] = pw[2];
        hg[t] = wmma_f16(aS, ub.v, zero);
      }
      float s[8];
#pragma unroll
      for (int r = 0; r < 8; ++r) s[r] = 0.f;
#pragma unroll
      for (int t = 0; t < 4; ++t) {
        float bb = sB1[mat * 64 + t * 16 + m16];
        float wc = sW2s[g * 64 + t * 16 + m16];
#pragma unroll
        for (int r = 0; r < 8; ++r)
          s[r] += softplus_f(hg[t][r] + bb) * wc;
      }
      // reduce over the 16 N-lanes of each half-wave; park result in lane g
#pragma unroll
      for (int r = 0; r < 8; ++r) {
        s[r] += __shfl_xor(s[r], 1, 32);
        s[r] += __shfl_xor(s[r], 2, 32);
        s[r] += __shfl_xor(s[r], 4, 32);
        s[r] += __shfl_xor(s[r], 8, 32);
        float v = s[r] + sB2s[g];
        keep[r] = (m16 == g) ? v : keep[r];
      }
    }

    if (m16 == 0) {
#pragma unroll
      for (int r = 0; r < 8; ++r) {
        int p = p0 + r + 8 * h2;
        __builtin_nontemporal_store(keep[r], o_sdf + p);
        __builtin_nontemporal_store(inv_d / (1.0f + __expf(keep[r] * inv_d)),
                                    o_sig + p);
      }
    } else if (m16 <= 4) {
#pragma unroll
      for (int r = 0; r < 8; ++r) {
        int p = p0 + r + 8 * h2;
        __builtin_nontemporal_store(keep[r], o_asdf + (size_t)p * 4 + (m16 - 1));
        __builtin_nontemporal_store(inv_d / (1.0f + __expf(keep[r] * inv_d)),
                                    o_asig + (size_t)p * 4 + (m16 - 1));
      }
    }
  }
}

extern "C" void kernel_launch(void* const* d_in, const int* in_sizes, int n_in,
                              void* d_out, int out_size, void* d_ws, size_t ws_size,
                              hipStream_t stream) {
  (void)in_sizes; (void)n_in; (void)out_size; (void)d_ws; (void)ws_size;
  const float* tex   = (const float*)d_in[0];
  const float* shp   = (const float*)d_in[1];
  // d_in[2] = ray_directions (unused by the reference computation)
  const float* Wtex1 = (const float*)d_in[3];
  const float* btex1 = (const float*)d_in[4];
  const float* Wtex2 = (const float*)d_in[5];
  const float* btex2 = (const float*)d_in[6];
  const float* Wshp1 = (const float*)d_in[7];
  const float* bshp1 = (const float*)d_in[8];
  const float* Wshp2 = (const float*)d_in[9];
  const float* bshp2 = (const float*)d_in[10];
  const float* Waux1 = (const float*)d_in[11];
  const float* baux1 = (const float*)d_in[12];
  const float* Waux2 = (const float*)d_in[13];
  const float* baux2 = (const float*)d_in[14];
  float* out = (float*)d_out;

  osg_decoder_kernel<<<dim3(BLOCKS), dim3(256), 0, stream>>>(
      tex, shp, Wtex1, btex1, Wtex2, btex2, Wshp1, bshp1, Wshp2, bshp2,
      Waux1, baux1, Waux2, baux2, out);
}
